// Qwen3Attention_86337432584459
// MI455X (gfx1250) — matrix operand order repaired and checked by reading, not compiled, not run on hardware
//
#include <hip/hip_runtime.h>
#include <hip/hip_bf16.h>

// ---- Qwen3 GQA attention constants ----
#define Bn   2
#define Sn   2048
#define HIDn 2048
#define NHn  16
#define NKVn 8
#define HDn  128
#define QKVO ((NHn + 2 * NKVn) * HDn)       // 4096
#define EPSc 1e-6f
#define SCALEc 0.08838834764831845f          // 128^-0.5

typedef __bf16 bf16;
typedef __attribute__((ext_vector_type(16))) __bf16 v16bf;
typedef __attribute__((ext_vector_type(8)))  __bf16 v8bf;
typedef __attribute__((ext_vector_type(8)))  float  v8f;

__device__ __forceinline__ v16bf join16(v8bf lo, v8bf hi) {
  union { v16bf v; v8bf h[2]; } u;
  u.h[0] = lo; u.h[1] = hi;
  return u.v;
}

__device__ __forceinline__ v8f wmma_bf16f32(v16bf a, v16bf b, v8f c) {
  // (neg_a, A, neg_b, B, c_mod, C, reuse_a, reuse_b)
  return __builtin_amdgcn_wmma_f32_16x16x32_bf16(false, a, false, b, (short)0, c,
                                                 false, false);
}

// Issue an async global->LDS 16-byte copy (per-lane), tracked by ASYNCcnt.
__device__ __forceinline__ void async_copy_b128(unsigned lds_off,
                                                unsigned long long gaddr) {
  asm volatile("global_load_async_to_lds_b128 %0, %1, off"
               :: "v"(lds_off), "v"(gaddr)
               : "memory");
}

// ---------------------------------------------------------------------------
// fp32 -> bf16 conversion
// ---------------------------------------------------------------------------
__global__ void __launch_bounds__(256)
f32_to_bf16(const float* __restrict__ in, bf16* __restrict__ out, int n) {
  int i = blockIdx.x * blockDim.x + threadIdx.x;
  const int stride = gridDim.x * blockDim.x;
  for (; i < n; i += stride) out[i] = (bf16)in[i];
}

// ---------------------------------------------------------------------------
// NT GEMM: C[M,N] = A[M,K] * B[N,K]^T   (bf16 in, fp32 WMMA accumulate).
// Block = 8 waves = 128(M) x 64(N) tile. The 64x32 B k-panel is shared by all
// 8 waves and is staged into LDS with GLOBAL_LOAD_ASYNC_TO_LDS_B128
// (256 thr x 16B = 4KB panel), double buffered: issue panel j+1, then
// s_wait_asynccnt 1 + barrier, and run 4 WMMAs per wave on panel j.
// A fragments are per-wave direct global loads (reused x4 per k-step).
// ---------------------------------------------------------------------------
template <typename OUT>
__global__ void __launch_bounds__(256)
gemm_nt_wmma(const bf16* __restrict__ A, const bf16* __restrict__ Bw,
             OUT* __restrict__ C, int M, int N, int K)
{
  __shared__ __align__(16) bf16 bbuf[2][64 * 32];   // 2 x 4KB B panels
  const int tid  = threadIdx.x;
  const int lane = tid & 31;
  const int wv   = tid >> 5;
  const int tiles_m = M >> 7;                        // 128-row block tiles
  const int m0 = (blockIdx.x % tiles_m) << 7;
  const int n0 = (blockIdx.x / tiles_m) << 6;
  const int half = lane >> 4;
  const int lr   = lane & 15;
  const int mw   = m0 + wv * 16;                     // this wave's 16 rows

  const bf16* arow = A + (size_t)(mw + lr) * K;

  // cooperative B-panel copy mapping: thread -> (row 0..63, 16B chunk 0..3)
  const int crow = tid >> 2;
  const int cchk = tid & 3;
  const bf16* gsrc = Bw + (size_t)(n0 + crow) * K + cchk * 8;
  const unsigned lbase0 = (unsigned)(uintptr_t)(&bbuf[0][0]) + crow * 64 + cchk * 16;
  const unsigned lbase1 = (unsigned)(uintptr_t)(&bbuf[1][0]) + crow * 64 + cchk * 16;

  const int niter = K >> 5;

  // prologue: stage k-panel 0 into buffer 0
  async_copy_b128(lbase0, (unsigned long long)(uintptr_t)gsrc);

  v8f acc[4] = {};
  for (int j = 0; j < niter; ++j) {
    const int k0 = j << 5;
    if (j + 1 < niter) {                             // wave-uniform branch
      async_copy_b128(((j + 1) & 1) ? lbase1 : lbase0,
                      (unsigned long long)(uintptr_t)(gsrc + k0 + 32));
      asm volatile("s_wait_asynccnt 0x1" ::: "memory");
    } else {
      asm volatile("s_wait_asynccnt 0x0" ::: "memory");
    }
    __syncthreads();                                 // panel j visible to all waves

    __builtin_prefetch(arow + k0 + 128, 0, 0);       // global_prefetch_b8 (A stream)
    v16bf af = join16(*(const v8bf*)(arow + k0 + half * 8),
                      *(const v8bf*)(arow + k0 + 16 + half * 8));
    const bf16* bp = &bbuf[j & 1][0];
#pragma unroll
    for (int t = 0; t < 4; ++t) {
      const bf16* br = bp + (t * 16 + lr) * 32 + half * 8;
      v16bf bfr = join16(*(const v8bf*)br, *(const v8bf*)(br + 16));
      acc[t] = wmma_bf16f32(af, bfr, acc[t]);
    }
    __syncthreads();                                 // done reading before j+2 overwrite
  }
#pragma unroll
  for (int t = 0; t < 4; ++t)
#pragma unroll
    for (int r = 0; r < 8; ++r) {
      const int m = mw + r + 8 * half;
      const int n = n0 + t * 16 + lr;
      C[(size_t)m * N + n] = (OUT)acc[t][r];
    }
}

// ---------------------------------------------------------------------------
// Per-head RMSNorm + RoPE + Q/K/V split. One wave per (b, s, head32).
// heads 0..15 -> Q (norm+rope), 16..23 -> K (norm+rope), 24..31 -> V (copy).
// V is written transposed [b, hkv, d, s] so flash PV B-fragments are contiguous.
// ---------------------------------------------------------------------------
__global__ void __launch_bounds__(256)
norm_rope_split(const bf16* __restrict__ qkv,
                const float* __restrict__ cosb, const float* __restrict__ sinb,
                const float* __restrict__ qw,   const float* __restrict__ kw,
                bf16* __restrict__ qout,        // [B, NH,  S, HD]
                bf16* __restrict__ kout,        // [B, NKV, S, HD]
                bf16* __restrict__ vtout)       // [B, NKV, HD, S]
{
  const int lane = threadIdx.x & 31;
  const int wid  = blockIdx.x * (blockDim.x >> 5) + (threadIdx.x >> 5);
  const int hh = wid & 31;
  const int s  = (wid >> 5) & (Sn - 1);
  const int b  = wid >> 16;               // / (32 * Sn)

  const bf16* src = qkv + (size_t)(b * Sn + s) * QKVO + hh * HDn;
  float x[4];
#pragma unroll
  for (int i = 0; i < 4; ++i) x[i] = (float)src[lane + 32 * i];

  if (hh >= 24) {                          // V head (wave-uniform branch)
    const int hv = hh - 24;
    bf16* dst = vtout + (size_t)(b * NKVn + hv) * HDn * Sn;
#pragma unroll
    for (int i = 0; i < 4; ++i) dst[(size_t)(lane + 32 * i) * Sn + s] = (bf16)x[i];
    return;
  }
  float ss = x[0] * x[0] + x[1] * x[1] + x[2] * x[2] + x[3] * x[3];
#pragma unroll
  for (int off = 1; off < 32; off <<= 1) ss += __shfl_xor(ss, off, 32);
  const float inv = rsqrtf(ss * (1.0f / HDn) + EPSc);
  const bool isq = (hh < 16);
  const float* w = isq ? qw : kw;
  float y[4];
#pragma unroll
  for (int i = 0; i < 4; ++i) y[i] = x[i] * inv * w[lane + 32 * i];
  float o[4];
#pragma unroll
  for (int i = 0; i < 2; ++i) {            // rope: d in [0,64) pairs with d+64
    const int d = lane + 32 * i;
    const float c  = cosb[s * (HDn / 2) + d];
    const float sn = sinb[s * (HDn / 2) + d];
    o[i]     = y[i] * c - y[i + 2] * sn;
    o[i + 2] = y[i + 2] * c + y[i] * sn;
  }
  bf16* dst = isq ? (qout + ((size_t)(b * NHn + hh) * Sn + s) * HDn)
                  : (kout + ((size_t)(b * NKVn + (hh - 16)) * Sn + s) * HDn);
#pragma unroll
  for (int i = 0; i < 4; ++i) dst[lane + 32 * i] = (bf16)o[i];
}

// ---------------------------------------------------------------------------
// Causal flash attention, one wave per 16-query tile, KV chunks of 32.
// QK^T: 8 WMMAs/chunk; online softmax (fp32 stats, shfl row reductions);
// P remapped C-layout -> A-layout through per-wave LDS (s_wait_dscnt);
// PV: 8 WMMAs/chunk against transposed V.
// ---------------------------------------------------------------------------
__global__ void __launch_bounds__(256)
flash_attn(const bf16* __restrict__ Q,   // [B, NH,  S, HD]
           const bf16* __restrict__ Kc,  // [B, NKV, S, HD]
           const bf16* __restrict__ Vt,  // [B, NKV, HD, S]
           bf16* __restrict__ O)         // [B, S, NH*HD]
{
  __shared__ __align__(16) bf16 pbuf[8][16 * 32];
  const int lane  = threadIdx.x & 31;
  const int wslot = threadIdx.x >> 5;
  const int wid   = blockIdx.x * 8 + wslot;   // B*NH*(S/16) waves total
  const int qt = wid & (Sn / 16 - 1);
  const int h  = (wid >> 7) & (NHn - 1);
  const int b  = wid >> 11;
  const int hk = h >> 1;                      // GQA: 2 query heads per KV head
  const int q0 = qt << 4;
  const int half = lane >> 4;
  const int lr   = lane & 15;

  const bf16* qrow = Q + ((size_t)(b * NHn + h) * Sn + q0 + lr) * HDn;
  v16bf qf[4];
#pragma unroll
  for (int dt = 0; dt < 4; ++dt) {
    const int d0 = dt * 32;
    qf[dt] = join16(*(const v8bf*)(qrow + d0 + half * 8),
                    *(const v8bf*)(qrow + d0 + 16 + half * 8));
  }

  v8f o[8] = {};
  float rowm[8], rowl[8];
#pragma unroll
  for (int r = 0; r < 8; ++r) { rowm[r] = -1e30f; rowl[r] = 0.0f; }

  const bf16* kbase = Kc + (size_t)(b * NKVn + hk) * Sn * HDn;
  const bf16* vbase = Vt + (size_t)(b * NKVn + hk) * HDn * Sn;
  bf16* pl = pbuf[wslot];

  const int nchunks = (q0 + 47) >> 5;          // kv chunks with any kv <= q_max
  for (int j = 0; j < nchunks; ++j) {
    const int kv0 = j << 5;
    v8f s1 = {}, s2 = {};
    const bf16* krow1 = kbase + (size_t)(kv0 + lr) * HDn + half * 8;
    const bf16* krow2 = krow1 + (size_t)16 * HDn;
#pragma unroll
    for (int dt = 0; dt < 4; ++dt) {
      v16bf k1 = join16(*(const v8bf*)(krow1 + dt * 32),
                        *(const v8bf*)(krow1 + dt * 32 + 16));
      v16bf k2 = join16(*(const v8bf*)(krow2 + dt * 32),
                        *(const v8bf*)(krow2 + dt * 32 + 16));
      s1 = wmma_bf16f32(qf[dt], k1, s1);
      s2 = wmma_bf16f32(qf[dt], k2, s2);
    }
    float p1[8], p2[8], alpha[8];
#pragma unroll
    for (int r = 0; r < 8; ++r) {
      const int q = q0 + r + 8 * half;
      float a = s1[r] * SCALEc; if (kv0 + lr > q)      a = -1e30f;
      float c = s2[r] * SCALEc; if (kv0 + 16 + lr > q) c = -1e30f;
      float t = fmaxf(a, c);
#pragma unroll
      for (int off = 1; off < 16; off <<= 1) t = fmaxf(t, __shfl_xor(t, off, 32));
      const float nm = fmaxf(rowm[r], t);
      alpha[r] = __expf(rowm[r] - nm);
      rowm[r] = nm;
      p1[r] = __expf(a - nm);
      p2[r] = __expf(c - nm);
      float rs = p1[r] + p2[r];
#pragma unroll
      for (int off = 1; off < 16; off <<= 1) rs += __shfl_xor(rs, off, 32);
      rowl[r] = rowl[r] * alpha[r] + rs;
    }
#pragma unroll
    for (int dt = 0; dt < 8; ++dt)
#pragma unroll
      for (int r = 0; r < 8; ++r) o[dt][r] *= alpha[r];

    // P: C/D layout -> A layout via per-wave LDS tile (16 rows x 32 kv, bf16)
#pragma unroll
    for (int r = 0; r < 8; ++r) {
      pl[(r + 8 * half) * 32 + lr]      = (bf16)p1[r];
      pl[(r + 8 * half) * 32 + 16 + lr] = (bf16)p2[r];
    }
    asm volatile("s_wait_dscnt 0" ::: "memory");
    const bf16* prow = pl + lr * 32;
    v16bf pf = join16(*(const v8bf*)(prow + half * 8),
                      *(const v8bf*)(prow + 16 + half * 8));
#pragma unroll
    for (int dt = 0; dt < 8; ++dt) {
      const bf16* vr = vbase + (size_t)(dt * 16 + lr) * Sn + kv0 + half * 8;
      v16bf vf = join16(*(const v8bf*)vr, *(const v8bf*)(vr + 16));
      o[dt] = wmma_bf16f32(pf, vf, o[dt]);
    }
  }
#pragma unroll
  for (int dt = 0; dt < 8; ++dt)
#pragma unroll
    for (int r = 0; r < 8; ++r) {
      const int q = q0 + r + 8 * half;
      O[(size_t)(b * Sn + q) * (NHn * HDn) + h * HDn + dt * 16 + lr] =
          (bf16)(o[dt][r] / rowl[r]);
    }
}

// ---------------------------------------------------------------------------
// Host-side launch pipeline (all on `stream`, graph-capture safe).
// ---------------------------------------------------------------------------
extern "C" void kernel_launch(void* const* d_in, const int* in_sizes, int n_in,
                              void* d_out, int out_size, void* d_ws, size_t ws_size,
                              hipStream_t stream)
{
  const float* hs   = (const float*)d_in[0];
  const float* cosb = (const float*)d_in[1];
  const float* sinb = (const float*)d_in[2];
  const float* wqkv = (const float*)d_in[3];
  const float* qnw  = (const float*)d_in[4];
  const float* knw  = (const float*)d_in[5];
  const float* wo   = (const float*)d_in[6];
  float* out = (float*)d_out;

  const size_t nHS   = (size_t)Bn * Sn * HIDn;        // 8.39M
  const size_t nWQKV = (size_t)QKVO * HIDn;           // 8.39M
  const size_t nWO   = (size_t)HIDn * NHn * HDn;      // 4.19M
  const size_t nQKV  = (size_t)Bn * Sn * QKVO;        // 16.8M
  const size_t nQ    = (size_t)Bn * NHn * Sn * HDn;   // 8.39M
  const size_t nKV   = (size_t)Bn * NKVn * Sn * HDn;  // 4.19M

  char* p = (char*)d_ws;
  bf16* hsb  = (bf16*)p; p += nHS   * sizeof(bf16);
  bf16* wqb  = (bf16*)p; p += nWQKV * sizeof(bf16);
  bf16* wob  = (bf16*)p; p += nWO   * sizeof(bf16);
  bf16* qkvb = (bf16*)p; p += nQKV  * sizeof(bf16);
  bf16* qb   = (bf16*)p; p += nQ    * sizeof(bf16);
  bf16* kb   = (bf16*)p; p += nKV   * sizeof(bf16);
  bf16* vtb  = (bf16*)p; p += nKV   * sizeof(bf16);
  bf16* attb = (bf16*)p; p += nQ    * sizeof(bf16);

  // 1) fp32 -> bf16 conversions
  f32_to_bf16<<<dim3((unsigned)((nHS + 255) / 256)), dim3(256), 0, stream>>>(hs, hsb, (int)nHS);
  f32_to_bf16<<<dim3((unsigned)((nWQKV + 255) / 256)), dim3(256), 0, stream>>>(wqkv, wqb, (int)nWQKV);
  f32_to_bf16<<<dim3((unsigned)((nWO + 255) / 256)), dim3(256), 0, stream>>>(wo, wob, (int)nWO);

  // 2) qkv = hs @ w_qkv^T  (M=4096, N=4096, K=2048), bf16 out
  {
    const int M = Bn * Sn, N = QKVO, K = HIDn;
    const int blocks = (M >> 7) * (N >> 6);          // 2048
    gemm_nt_wmma<bf16><<<dim3(blocks), dim3(256), 0, stream>>>(hsb, wqb, qkvb, M, N, K);
  }
  // 3) RMSNorm + RoPE + split (V transposed)
  {
    const int waves = Bn * Sn * 32;                  // 131072
    norm_rope_split<<<dim3(waves / 8), dim3(256), 0, stream>>>(qkvb, cosb, sinb,
                                                               qnw, knw, qb, kb, vtb);
  }
  // 4) causal flash attention
  {
    const int waves = Bn * NHn * (Sn / 16);          // 4096
    flash_attn<<<dim3(waves / 8), dim3(256), 0, stream>>>(qb, kb, vtb, attb);
  }
  // 5) out = attn @ w_o^T  (M=4096, N=2048, K=2048), fp32 out -> d_out
  {
    const int M = Bn * Sn, N = HIDn, K = NHn * HDn;
    const int blocks = (M >> 7) * (N >> 6);          // 1024
    gemm_nt_wmma<float><<<dim3(blocks), dim3(256), 0, stream>>>(attb, wob, out, M, N, K);
  }
  (void)in_sizes; (void)n_in; (void)out_size; (void)ws_size;
}
